// MultiHeadAttention_6614249635865
// MI455X (gfx1250) — matrix operand order repaired and checked by reading, not compiled, not run on hardware
//
#include <hip/hip_runtime.h>
#include <hip/hip_bf16.h>

// ---------------------------------------------------------------------------
// MHA forward for MI455X (gfx1250), wave32, bf16 WMMA w/ fp32 accumulation.
//   B=2, T=2048, D=1024, H=16, Hd=64, SCALE = 1/8
// Pipeline: [qkv proj GEMM] -> [flash attention] -> [out proj GEMM]
// ---------------------------------------------------------------------------

typedef __bf16 bf_t;
typedef __attribute__((ext_vector_type(16))) __bf16 v16bf;
typedef __attribute__((ext_vector_type(8)))  __bf16 v8bf;
typedef __attribute__((ext_vector_type(8)))  float  v8f;
typedef int vi4 __attribute__((vector_size(4 * sizeof(int))));

#if defined(__has_builtin)
#if __has_builtin(__builtin_amdgcn_global_load_async_to_lds_b128)
#define ASYNC_LDS 1
#endif
#endif

static __device__ __forceinline__ v8bf cvt8(const float* p) {
  const float4 a = *reinterpret_cast<const float4*>(p);
  const float4 b = *reinterpret_cast<const float4*>(p + 4);
  v8bf r;
  r[0] = (__bf16)a.x; r[1] = (__bf16)a.y; r[2] = (__bf16)a.z; r[3] = (__bf16)a.w;
  r[4] = (__bf16)b.x; r[5] = (__bf16)b.y; r[6] = (__bf16)b.z; r[7] = (__bf16)b.w;
  return r;
}
static __device__ __forceinline__ v8bf load8(const float* p)  { return cvt8(p); }
static __device__ __forceinline__ v8bf load8(const __bf16* p) { return *reinterpret_cast<const v8bf*>(p); }

static __device__ __forceinline__ v16bf cat16(v8bf lo, v8bf hi) {
  return __builtin_shufflevector(lo, hi, 0,1,2,3,4,5,6,7,8,9,10,11,12,13,14,15);
}

static __device__ __forceinline__ v8f wmma_bf(v16bf a, v16bf b, v8f c) {
  // D = A(16x32 bf16) x B(32x16 bf16) + C(16x16 f32)
  return __builtin_amdgcn_wmma_f32_16x16x32_bf16(false, a, false, b, (short)0, c, false, false);
}

// 16-byte global -> LDS copy; async (ASYNCcnt-tracked) when the toolchain
// exposes the gfx1250 builtin, otherwise a plain b128 round trip.
static __device__ __forceinline__ void stage16(bf_t* lds_dst, const bf_t* g_src) {
#ifdef ASYNC_LDS
  __builtin_amdgcn_global_load_async_to_lds_b128(
      (__attribute__((address_space(1))) vi4*)g_src,
      (__attribute__((address_space(3))) vi4*)lds_dst, 0, 0);
#else
  *reinterpret_cast<v8bf*>(lds_dst) = *reinterpret_cast<const v8bf*>(g_src);
#endif
}
static __device__ __forceinline__ void wait_async() {
#ifdef ASYNC_LDS
#if __has_builtin(__builtin_amdgcn_s_wait_asynccnt)
  __builtin_amdgcn_s_wait_asynccnt(0);
#else
  asm volatile("s_wait_asynccnt 0x0" ::: "memory");
#endif
#endif
}

// ---------------------------------------------------------------------------
// Kernel 1: fused QKV projection.  C[m,o] = X[m,:] . W[o,:] + bias[o]
// Output written bf16 in [b][h][t][hd] layout; SCALE folded into Q (z==0).
// Block: 256 threads (8 waves); block tile 256(M) x 64(N); wave tile 32x64.
// ---------------------------------------------------------------------------
__global__ __launch_bounds__(256) void qkv_proj_kernel(
    const float* __restrict__ X,
    const float* __restrict__ wq, const float* __restrict__ wk, const float* __restrict__ wv,
    const float* __restrict__ bq, const float* __restrict__ bk, const float* __restrict__ bv,
    bf_t* __restrict__ qb, bf_t* __restrict__ kb, bf_t* __restrict__ vb)
{
  constexpr int Dm = 1024, T = 2048, H = 16;
  const int z = blockIdx.z;
  const float* W    = (z == 0) ? wq : ((z == 1) ? wk : wv);
  const float* bias = (z == 0) ? bq : ((z == 1) ? bk : bv);
  bf_t* outp        = (z == 0) ? qb : ((z == 1) ? kb : vb);
  const float outscale = (z == 0) ? 0.125f : 1.0f;   // SCALE = Hd^-0.5 = 1/8

  __shared__ alignas(16) bf_t lds_w[64][32];         // W tile [n][k], bf16

  const int tid  = threadIdx.x;
  const int wave = tid >> 5, lane = tid & 31, l16 = lane & 15, hi = lane >> 4;
  const int m0 = blockIdx.y * 256 + wave * 32;       // 32 rows per wave
  const int n0 = blockIdx.x * 64;
  const int koffA = hi ? 8 : 0;
  const int koffB = hi ? 8 : 0;

  v8f acc[2][4] = {};
  const int wn  = tid >> 2;        // 0..63
  const int wk8 = (tid & 3) * 8;   // 0,8,16,24

  for (int k0 = 0; k0 < Dm; k0 += 32) {
    *reinterpret_cast<v8bf*>(&lds_w[wn][wk8]) = cvt8(&W[(size_t)(n0 + wn) * Dm + k0 + wk8]);
    __syncthreads();

    v16bf afrag[2];
#pragma unroll
    for (int t = 0; t < 2; ++t) {
      const float* ap = &X[(size_t)(m0 + t * 16 + l16) * Dm + k0 + koffA];
      afrag[t] = cat16(load8(ap), load8(ap + 16));
    }
#pragma unroll
    for (int c = 0; c < 4; ++c) {
      v16bf bfrag = cat16(*reinterpret_cast<const v8bf*>(&lds_w[c * 16 + l16][koffB]),
                          *reinterpret_cast<const v8bf*>(&lds_w[c * 16 + l16][koffB + 16]));
      acc[0][c] = wmma_bf(afrag[0], bfrag, acc[0][c]);
      acc[1][c] = wmma_bf(afrag[1], bfrag, acc[1][c]);
    }
    __syncthreads();
  }

#pragma unroll
  for (int t = 0; t < 2; ++t) {
#pragma unroll
    for (int c = 0; c < 4; ++c) {
      const int o = n0 + c * 16 + l16;
      const int h = o >> 6, hd = o & 63;
      const float bval = bias[o];
#pragma unroll
      for (int r = 0; r < 8; ++r) {
        const int m = m0 + t * 16 + r + hi * 8;
        const int b = m >> 11, tt = m & 2047;  // T = 2048
        outp[((size_t)(b * H + h) * T + tt) * 64 + hd] =
            (bf_t)((acc[t][c][r] + bval) * outscale);
      }
    }
  }
}

// ---------------------------------------------------------------------------
// Kernel 2: flash attention over one (b,h) per blockIdx.y.
// Block: 8 waves x 16 query rows = 128 queries; key loop in chunks of 64.
// K chunk staged async into LDS [key][d]; V chunk transposed to [d][key].
// Online softmax stats live in the lanes that own the C-layout rows.
// ---------------------------------------------------------------------------
__global__ __launch_bounds__(256) void flash_attn_kernel(
    const bf_t* __restrict__ qb, const bf_t* __restrict__ kb,
    const bf_t* __restrict__ vb, bf_t* __restrict__ attnb)
{
  constexpr int T = 2048, Hd = 64;
  __shared__ alignas(16) bf_t lds_k[64][64];       // K chunk [key][d]     (8 KB)
  __shared__ alignas(16) bf_t lds_vt[64][64];      // V chunk [d][key]     (8 KB)
  __shared__ alignas(16) bf_t lds_p[8][16][64];    // per-wave P tiles     (16 KB)

  const int tid  = threadIdx.x;
  const int wave = tid >> 5, lane = tid & 31, l16 = lane & 15, hi = lane >> 4;
  const int koffA = hi ? 8 : 0, koffB = hi ? 8 : 0;
  const int bh = blockIdx.y;
  const bf_t* qh = qb + (size_t)bh * T * Hd;
  const bf_t* kh = kb + (size_t)bh * T * Hd;
  const bf_t* vh = vb + (size_t)bh * T * Hd;
  const int qt0 = blockIdx.x * 128 + wave * 16;

  // Q fragments (SCALE already folded in): two K=32 steps covering Hd=64
  v16bf qfrag[2];
#pragma unroll
  for (int kk = 0; kk < 2; ++kk) {
    const bf_t* ap = &qh[(size_t)(qt0 + l16) * Hd + kk * 32 + koffA];
    qfrag[kk] = cat16(load8(ap), load8(ap + 16));
  }

  float m_i[8], l_i[8];
#pragma unroll
  for (int r = 0; r < 8; ++r) { m_i[r] = -1e30f; l_i[r] = 0.0f; }
  v8f oacc[4] = {};

  const int srow = tid >> 2;        // 0..63 (key row within chunk)
  const int sc16 = (tid & 3) * 16;  // 0,16,32,48 (dim offset)

  for (int j0 = 0; j0 < T; j0 += 64) {
    // stage K chunk (row-major, async) and V chunk (transposed, via regs)
    stage16(&lds_k[srow][sc16],     &kh[(size_t)(j0 + srow) * Hd + sc16]);
    stage16(&lds_k[srow][sc16 + 8], &kh[(size_t)(j0 + srow) * Hd + sc16 + 8]);
    v8bf v0 = *reinterpret_cast<const v8bf*>(&vh[(size_t)(j0 + srow) * Hd + sc16]);
    v8bf v1 = *reinterpret_cast<const v8bf*>(&vh[(size_t)(j0 + srow) * Hd + sc16 + 8]);
#pragma unroll
    for (int i = 0; i < 8; ++i) {
      lds_vt[sc16 + i][srow]     = v0[i];
      lds_vt[sc16 + 8 + i][srow] = v1[i];
    }
    wait_async();
    __syncthreads();

    // S(16x64) = Q(16x64) . K^T : 4 column tiles x 2 K-steps = 8 WMMAs
    v8f s[4] = {};
#pragma unroll
    for (int c = 0; c < 4; ++c) {
#pragma unroll
      for (int kk = 0; kk < 2; ++kk) {
        v16bf bk = cat16(
            *reinterpret_cast<const v8bf*>(&lds_k[c * 16 + l16][kk * 32 + koffB]),
            *reinterpret_cast<const v8bf*>(&lds_k[c * 16 + l16][kk * 32 + koffB + 16]));
        s[c] = wmma_bf(qfrag[kk], bk, s[c]);
      }
    }

    // online softmax: row reductions across the 16-lane half-wave groups
    float alpha[8];
#pragma unroll
    for (int r = 0; r < 8; ++r) {
      float sm = fmaxf(fmaxf(s[0][r], s[1][r]), fmaxf(s[2][r], s[3][r]));
      sm = fmaxf(sm, __shfl_xor(sm, 1, 32));
      sm = fmaxf(sm, __shfl_xor(sm, 2, 32));
      sm = fmaxf(sm, __shfl_xor(sm, 4, 32));
      sm = fmaxf(sm, __shfl_xor(sm, 8, 32));
      const float mnew = fmaxf(m_i[r], sm);
      alpha[r] = __expf(m_i[r] - mnew);
      float p0 = __expf(s[0][r] - mnew);
      float p1 = __expf(s[1][r] - mnew);
      float p2 = __expf(s[2][r] - mnew);
      float p3 = __expf(s[3][r] - mnew);
      float rs = (p0 + p1) + (p2 + p3);
      rs += __shfl_xor(rs, 1, 32);
      rs += __shfl_xor(rs, 2, 32);
      rs += __shfl_xor(rs, 4, 32);
      rs += __shfl_xor(rs, 8, 32);
      l_i[r] = l_i[r] * alpha[r] + rs;
      m_i[r] = mnew;
      s[0][r] = p0; s[1][r] = p1; s[2][r] = p2; s[3][r] = p3;
    }
#pragma unroll
    for (int c = 0; c < 4; ++c)
#pragma unroll
      for (int r = 0; r < 8; ++r) oacc[c][r] *= alpha[r];

    // relayout P (C-layout, f32) -> A-fragments (bf16) via per-wave LDS tile
#pragma unroll
    for (int r = 0; r < 8; ++r) {
#pragma unroll
      for (int c = 0; c < 4; ++c)
        lds_p[wave][r + hi * 8][c * 16 + l16] = (bf_t)s[c][r];
    }
    v16bf pfrag[2];
#pragma unroll
    for (int kk = 0; kk < 2; ++kk) {
      const bf_t* pp = &lds_p[wave][l16][kk * 32 + koffA];
      pfrag[kk] = cat16(*reinterpret_cast<const v8bf*>(pp),
                        *reinterpret_cast<const v8bf*>(pp + 16));
    }

    // O(16x64) += P(16x64) . V(64x64) : 4 dim tiles x 2 K-steps = 8 WMMAs
#pragma unroll
    for (int c = 0; c < 4; ++c) {
#pragma unroll
      for (int kk = 0; kk < 2; ++kk) {
        v16bf bvf = cat16(
            *reinterpret_cast<const v8bf*>(&lds_vt[c * 16 + l16][kk * 32 + koffB]),
            *reinterpret_cast<const v8bf*>(&lds_vt[c * 16 + l16][kk * 32 + koffB + 16]));
        oacc[c] = wmma_bf(pfrag[kk], bvf, oacc[c]);
      }
    }
    __syncthreads();
  }

  // epilogue: O /= l ; write bf16 in [b][t][h*64+hd] layout for out-proj
  const int b = bh >> 4, h = bh & 15;
#pragma unroll
  for (int c = 0; c < 4; ++c) {
#pragma unroll
    for (int r = 0; r < 8; ++r) {
      const int t = qt0 + r + hi * 8;
      attnb[((size_t)(b * T + t)) * 1024 + h * 64 + c * 16 + l16] =
          (bf_t)(oacc[c][r] / l_i[r]);
    }
  }
}

// ---------------------------------------------------------------------------
// Kernel 3: output projection.  out[m,o] = A[m,:] . Wo[o,:] + bo[o]  (fp32 out)
// Block tile 256(M) x 64(N); wave tile 32x64.
// ---------------------------------------------------------------------------
__global__ __launch_bounds__(256) void out_proj_kernel(
    const bf_t* __restrict__ A, const float* __restrict__ W,
    const float* __restrict__ bias, float* __restrict__ Cout)
{
  constexpr int Dm = 1024;
  __shared__ alignas(16) bf_t lds_w[64][32];

  const int tid  = threadIdx.x;
  const int wave = tid >> 5, lane = tid & 31, l16 = lane & 15, hi = lane >> 4;
  const int m0 = blockIdx.y * 256 + wave * 32;
  const int n0 = blockIdx.x * 64;
  const int koffA = hi ? 8 : 0;
  const int koffB = hi ? 8 : 0;

  v8f acc[2][4] = {};
  const int wn  = tid >> 2;
  const int wk8 = (tid & 3) * 8;

  for (int k0 = 0; k0 < Dm; k0 += 32) {
    *reinterpret_cast<v8bf*>(&lds_w[wn][wk8]) = cvt8(&W[(size_t)(n0 + wn) * Dm + k0 + wk8]);
    __syncthreads();

    v16bf afrag[2];
#pragma unroll
    for (int t = 0; t < 2; ++t) {
      const bf_t* ap = &A[(size_t)(m0 + t * 16 + l16) * Dm + k0 + koffA];
      afrag[t] = cat16(load8(ap), load8(ap + 16));
    }
#pragma unroll
    for (int c = 0; c < 4; ++c) {
      v16bf bfrag = cat16(*reinterpret_cast<const v8bf*>(&lds_w[c * 16 + l16][koffB]),
                          *reinterpret_cast<const v8bf*>(&lds_w[c * 16 + l16][koffB + 16]));
      acc[0][c] = wmma_bf(afrag[0], bfrag, acc[0][c]);
      acc[1][c] = wmma_bf(afrag[1], bfrag, acc[1][c]);
    }
    __syncthreads();
  }

#pragma unroll
  for (int t = 0; t < 2; ++t) {
#pragma unroll
    for (int c = 0; c < 4; ++c) {
      const int o = n0 + c * 16 + l16;
      const float bval = bias[o];
#pragma unroll
      for (int r = 0; r < 8; ++r) {
        const int m = m0 + t * 16 + r + hi * 8;
        Cout[(size_t)m * 1024 + o] = acc[t][c][r] + bval;
      }
    }
  }
}

// ---------------------------------------------------------------------------
extern "C" void kernel_launch(void* const* d_in, const int* in_sizes, int n_in,
                              void* d_out, int out_size, void* d_ws, size_t ws_size,
                              hipStream_t stream) {
  const float* query = (const float*)d_in[0];
  const float* w_q   = (const float*)d_in[1];
  const float* w_k   = (const float*)d_in[2];
  const float* w_v   = (const float*)d_in[3];
  const float* w_o   = (const float*)d_in[4];
  const float* b_q   = (const float*)d_in[5];
  const float* b_k   = (const float*)d_in[6];
  const float* b_v   = (const float*)d_in[7];
  const float* b_o   = (const float*)d_in[8];
  float* out = (float*)d_out;

  constexpr size_t SZ = (size_t)2 * 2048 * 1024;   // elements per bf16 tensor
  bf_t* ws    = (bf_t*)d_ws;
  bf_t* qb    = ws;            // [b][h][t][hd]
  bf_t* kb    = ws + SZ;
  bf_t* vb    = ws + 2 * SZ;
  bf_t* attnb = ws + 3 * SZ;   // [b][t][h*hd]

  const dim3 blk(256);
  // M = B*T = 4096 rows, N = 1024 cols -> grid (N/64, M/256)
  qkv_proj_kernel<<<dim3(16, 16, 3), blk, 0, stream>>>(
      query, w_q, w_k, w_v, b_q, b_k, b_v, qb, kb, vb);
  // T/128 = 16 query blocks, B*H = 32 heads
  flash_attn_kernel<<<dim3(16, 32), blk, 0, stream>>>(qb, kb, vb, attnb);
  out_proj_kernel<<<dim3(16, 16), blk, 0, stream>>>(attnb, w_o, b_o, out);
}
